// MultiQueryAttention_12807592477441
// MI455X (gfx1250) — matrix operand order repaired and checked by reading, not compiled, not run on hardware
//
#include <hip/hip_runtime.h>
#include <cstdint>
#include <cstddef>

// ---------------------------------------------------------------------------
// CDNA5 (gfx1250) MQA forward, bf16 WMMA (v_wmma_f32_16x16x32_bf16) everywhere,
// with async global->LDS tile staging (ASYNCcnt), software-pipelined fragment
// loads (DS/WMMA overlap), and global prefetch.
// ---------------------------------------------------------------------------

typedef __attribute__((ext_vector_type(16))) __bf16 bf16x16;
typedef __attribute__((ext_vector_type(8)))  float  f32x8;

union Frag16 {
  uint32_t d[8];
  uint4    q[2];
  bf16x16  v;
};

__device__ __forceinline__ uint16_t f2bf(float f) {
  uint32_t u = __float_as_uint(f);
  u += 0x7FFFu + ((u >> 16) & 1u);   // round-to-nearest-even
  return (uint16_t)(u >> 16);
}

__device__ __forceinline__ f32x8 wmma_bf16(const Frag16& a, const Frag16& b, f32x8 c) {
  // 8 args: (neg_a, A, neg_b, B, c_mod, C, reuse_a, reuse_b)
  return __builtin_amdgcn_wmma_f32_16x16x32_bf16(false, a.v, false, b.v,
                                                 (short)0, c, false, false);
}

// async global -> LDS copy of 16 bytes (per-lane), tracked by ASYNCcnt
__device__ __forceinline__ void async_b128_to_lds(const void* gaddr, void* lds_flat) {
  uint32_t lds = (uint32_t)(uintptr_t)lds_flat;  // LDS aperture: addr[31:0] = LDS offset
  asm volatile("global_load_async_to_lds_b128 %0, %1, off"
               :: "v"(lds), "v"(gaddr) : "memory");
}

__device__ __forceinline__ void wait_asynccnt0() {
  asm volatile("s_wait_asynccnt 0" ::: "memory");
}

// ---------------------------------------------------------------------------
// fp32 -> bf16 conversion (grid-stride)
// ---------------------------------------------------------------------------
__global__ void cvt_f32_to_bf16(const float* __restrict__ in,
                                uint16_t* __restrict__ out, size_t n) {
  size_t i = (size_t)blockIdx.x * blockDim.x + threadIdx.x;
  size_t stride = (size_t)gridDim.x * blockDim.x;
  for (; i < n; i += stride) out[i] = f2bf(in[i]);
}

// ---------------------------------------------------------------------------
// GEMM: out[M,N] = A[M,K](bf16) @ W[N,K]^T(bf16) + bias[N]
// Each wave computes a (TM*16) x (TN*16) tile. K iterated in chunks of 32.
// B fragments are double-buffered across the TN loop for load/WMMA overlap.
// TRANS: store transposed bf16 as out[b][n][t] (for V^T), ldT = rows per batch.
// F32OUT: store fp32 (final output), else bf16.
// ---------------------------------------------------------------------------
template <int TM, int TN, bool TRANS, bool F32OUT>
__global__ __launch_bounds__(256) void gemm_bf16_wmma(
    const uint16_t* __restrict__ A, const uint16_t* __restrict__ W,
    const float* __restrict__ bias, void* __restrict__ outp,
    int M, int N, int K, int ldT) {
  const int lane = threadIdx.x & 31;
  const int wid  = blockIdx.x * (blockDim.x >> 5) + (threadIdx.x >> 5);
  const int tilesN = N / (TN * 16);
  const int tilesM = M / (TM * 16);
  if (wid >= tilesM * tilesN) return;
  const int tm0 = (wid / tilesN) * (TM * 16);
  const int tn0 = (wid % tilesN) * (TN * 16);
  const int lr  = lane & 15;
  const bool hi = lane >= 16;

  f32x8 acc[TM][TN];
#pragma unroll
  for (int i = 0; i < TM; ++i)
#pragma unroll
    for (int j = 0; j < TN; ++j)
#pragma unroll
      for (int r = 0; r < 8; ++r) acc[i][j][r] = 0.f;

  for (int k0 = 0; k0 < K; k0 += 32) {
    if (k0 + 32 < K) {  // prefetch next K-chunk (global_prefetch_b8)
      __builtin_prefetch(A + (size_t)(tm0 + lr) * K + k0 + 32, 0, 1);
      __builtin_prefetch(W + (size_t)(tn0 + lr) * K + k0 + 32, 0, 1);
    }
    Frag16 af[TM];
#pragma unroll
    for (int tm = 0; tm < TM; ++tm) {
      // A-layout: lanes 0-15 -> M=lane, halves K {0..7,16..23}; lanes 16-31 -> {8..15,24..31}
      const uint16_t* p = A + (size_t)(tm0 + tm * 16 + lr) * K + k0 + (hi ? 8 : 0);
      af[tm].q[0] = *(const uint4*)(p);
      af[tm].q[1] = *(const uint4*)(p + 16);
    }
    // B-layout: lanes 0-15 -> N=lane, K {0..7,16..23}; lanes 16-31 -> N=lane-16, K {8..15,24..31}
    auto loadB = [&](Frag16& fr, int tn) {
      const uint16_t* p = W + (size_t)(tn0 + tn * 16 + lr) * K + k0 + (hi ? 8 : 0);
      fr.q[0] = *(const uint4*)(p);
      fr.q[1] = *(const uint4*)(p + 16);
    };
    Frag16 bcur;
    loadB(bcur, 0);
#pragma unroll
    for (int tn = 0; tn < TN; ++tn) {
      Frag16 bnext;
      if (tn + 1 < TN) loadB(bnext, tn + 1);   // overlap next loads with WMMAs
#pragma unroll
      for (int tm = 0; tm < TM; ++tm)
        acc[tm][tn] = wmma_bf16(af[tm], bcur, acc[tm][tn]);
      if (tn + 1 < TN) bcur = bnext;
    }
  }

#pragma unroll
  for (int tm = 0; tm < TM; ++tm) {
#pragma unroll
    for (int tn = 0; tn < TN; ++tn) {
      const int n = tn0 + tn * 16 + lr;
      const float bv = bias[n];
      const int mrow = tm0 + tm * 16 + (hi ? 8 : 0);  // D-layout: rows 0-7 / 8-15
      if (!TRANS) {
        if (F32OUT) {
          float* out = (float*)outp;
#pragma unroll
          for (int r = 0; r < 8; ++r)
            out[(size_t)(mrow + r) * N + n] = acc[tm][tn][r] + bv;
        } else {
          uint16_t* out = (uint16_t*)outp;
#pragma unroll
          for (int r = 0; r < 8; ++r)
            out[(size_t)(mrow + r) * N + n] = f2bf(acc[tm][tn][r] + bv);
        }
      } else {
        // V^T store: lane holds 8 consecutive t values of one d column -> 16B store
        uint16_t* out = (uint16_t*)outp;
        const int bidx = mrow / ldT;
        const int t = mrow % ldT;
        uint32_t pk[4];
#pragma unroll
        for (int r = 0; r < 4; ++r)
          pk[r] = (uint32_t)f2bf(acc[tm][tn][2 * r] + bv) |
                  ((uint32_t)f2bf(acc[tm][tn][2 * r + 1] + bv) << 16);
        *(uint4*)(out + ((size_t)bidx * N + n) * ldT + t) =
            make_uint4(pk[0], pk[1], pk[2], pk[3]);
      }
    }
  }
}

// ---------------------------------------------------------------------------
// Fused flash-attention (MQA): one block per (b, h, 128-query tile).
// 8 waves x 16 query rows. K/V tiles of 32 keys staged in LDS via
// global_load_async_to_lds_b128, shared by all waves (single KV head).
// Online softmax, P transposed via per-wave LDS. Fragment loads are
// software-pipelined one step ahead of their consuming WMMA.
// ---------------------------------------------------------------------------
#define ATT_SCALE 0.022097086912079608f  // 1/sqrt(2048)

__global__ __launch_bounds__(256) void mqa_attn(
    const uint16_t* __restrict__ Qb,  // [B*S, E] bf16
    const uint16_t* __restrict__ Kb,  // [B*S, D] bf16
    const uint16_t* __restrict__ Vt,  // [B, D, S] bf16 (V transposed)
    uint16_t* __restrict__ Ob) {      // [B*S, E] bf16
  constexpr int S = 2048, E = 2048, D = 128, H = 16;
  __shared__ __align__(16) uint16_t Ktile[32][128];   // [t][d]  8 KB
  __shared__ __align__(16) uint16_t Vtile[128][32];   // [d][t]  8 KB
  __shared__ __align__(16) uint16_t Pbuf[8][32][16];  // [wave][t][q] 8 KB

  const int lane = threadIdx.x & 31;
  const int wave = threadIdx.x >> 5;
  const int lr = lane & 15;
  const bool hi = lane >= 16;

  const int nQT = S / 128;  // 16
  const int qt = blockIdx.x % nQT;
  const int bh = blockIdx.x / nQT;
  const int b = bh / H;
  const int h = bh % H;
  const int q0 = qt * 128 + wave * 16;

  // Q fragments: 16 rows x 128 d -> 4 A-fragments (16x32)
  Frag16 qf[4];
  {
    const uint16_t* qrow = Qb + (size_t)(b * S + q0 + lr) * E + h * D;
#pragma unroll
    for (int kk = 0; kk < 4; ++kk) {
      const uint16_t* p = qrow + kk * 32 + (hi ? 8 : 0);
      qf[kk].q[0] = *(const uint4*)(p);
      qf[kk].q[1] = *(const uint4*)(p + 16);
    }
  }

  // per-thread cooperative tile-copy coordinates
  const int kr = threadIdx.x >> 3;            // 0..31 (key row)
  const int kc = (threadIdx.x & 7) * 16;      // 0..112 (d col)
  const int vr = threadIdx.x >> 1;            // 0..127 (d row)
  const int vc = (threadIdx.x & 1) * 16;      // 0/16  (t col)
  const uint16_t* gK0 = Kb + (size_t)(b * S + kr) * D + kc;
  const uint16_t* gV0 = Vt + (size_t)(b * D + vr) * S + vc;

  f32x8 acc[8];
#pragma unroll
  for (int f = 0; f < 8; ++f)
#pragma unroll
    for (int r = 0; r < 8; ++r) acc[f][r] = 0.f;
  float mrow[8], lrow[8];
#pragma unroll
  for (int r = 0; r < 8; ++r) { mrow[r] = -3.0e38f; lrow[r] = 0.f; }

  for (int t0 = 0; t0 < S; t0 += 32) {
    __syncthreads();  // all waves done reading previous tiles
    {
      // async global -> LDS (one b128 per thread per tile), ASYNCcnt-tracked
      async_b128_to_lds(gK0 + (size_t)t0 * D, &Ktile[kr][kc]);
      async_b128_to_lds(gV0 + t0, &Vtile[vr][vc]);
      if (t0 + 32 < S) {  // prefetch next tiles into cache
        __builtin_prefetch(gK0 + (size_t)(t0 + 32) * D, 0, 1);
        __builtin_prefetch(gV0 + t0 + 32, 0, 1);
      }
    }
    wait_asynccnt0();  // LDS writes landed (this wave's async ops)
    __syncthreads();   // publish tiles to all waves

    // scores: S[16q x 32t], two 16-key subtiles, K-fragments pipelined
    auto loadK = [&](Frag16& fr, int st, int kk) {
      const uint16_t* p = &Ktile[st * 16 + lr][kk * 32 + (hi ? 8 : 0)];
      fr.q[0] = *(const uint4*)(p);
      fr.q[1] = *(const uint4*)(p + 16);
    };
    f32x8 c0, c1;
#pragma unroll
    for (int r = 0; r < 8; ++r) { c0[r] = 0.f; c1[r] = 0.f; }
    Frag16 kb0, kb1;
    loadK(kb0, 0, 0);
    loadK(kb1, 1, 0);
#pragma unroll
    for (int kk = 0; kk < 4; ++kk) {
      Frag16 kn0, kn1;
      if (kk + 1 < 4) { loadK(kn0, 0, kk + 1); loadK(kn1, 1, kk + 1); }
      c0 = wmma_bf16(qf[kk], kb0, c0);
      c1 = wmma_bf16(qf[kk], kb1, c1);
      if (kk + 1 < 4) { kb0 = kn0; kb1 = kn1; }
    }
    f32x8 s[2];
#pragma unroll
    for (int r = 0; r < 8; ++r) { s[0][r] = c0[r] * ATT_SCALE; s[1][r] = c1[r] * ATT_SCALE; }

    // online softmax: row r lives in VGPR r across a 16-lane group
    float cm[8];
#pragma unroll
    for (int r = 0; r < 8; ++r) {
      float v = fmaxf(s[0][r], s[1][r]);
      v = fmaxf(v, __shfl_xor(v, 1));
      v = fmaxf(v, __shfl_xor(v, 2));
      v = fmaxf(v, __shfl_xor(v, 4));
      v = fmaxf(v, __shfl_xor(v, 8));
      cm[r] = v;
    }
    float resc[8];
#pragma unroll
    for (int r = 0; r < 8; ++r) {
      const float mn = fmaxf(mrow[r], cm[r]);
      resc[r] = __expf(mrow[r] - mn);
      mrow[r] = mn;
    }
#pragma unroll
    for (int st = 0; st < 2; ++st)
#pragma unroll
      for (int r = 0; r < 8; ++r) s[st][r] = __expf(s[st][r] - mrow[r]);
#pragma unroll
    for (int r = 0; r < 8; ++r) {
      float v = s[0][r] + s[1][r];
      v += __shfl_xor(v, 1);
      v += __shfl_xor(v, 2);
      v += __shfl_xor(v, 4);
      v += __shfl_xor(v, 8);
      lrow[r] = lrow[r] * resc[r] + v;
    }
#pragma unroll
    for (int f = 0; f < 8; ++f)
#pragma unroll
      for (int r = 0; r < 8; ++r) acc[f][r] *= resc[r];

    // P (C-layout) -> per-wave LDS as [t][q] -> reload in A-layout
#pragma unroll
    for (int st = 0; st < 2; ++st) {
      uint32_t pk[4];
#pragma unroll
      for (int r = 0; r < 4; ++r)
        pk[r] = (uint32_t)f2bf(s[st][2 * r]) | ((uint32_t)f2bf(s[st][2 * r + 1]) << 16);
      *(uint4*)&Pbuf[wave][st * 16 + lr][hi ? 8 : 0] =
          make_uint4(pk[0], pk[1], pk[2], pk[3]);
    }
    asm volatile("s_wait_dscnt 0" ::: "memory");  // cross-lane LDS dep, same wave

    Frag16 pf;  // A-fragment: 16 q rows x 32 t
    {
      uint16_t u[16];
      const int tb = hi ? 8 : 0;
#pragma unroll
      for (int i = 0; i < 16; ++i) {
        const int t = tb + (i & 7) + ((i >= 8) ? 16 : 0);
        u[i] = Pbuf[wave][t][lr];
      }
#pragma unroll
      for (int i = 0; i < 8; ++i)
        pf.d[i] = (uint32_t)u[2 * i] | ((uint32_t)u[2 * i + 1] << 16);
    }

    // O += P @ V : 8 d-tiles, V-fragments pipelined one ahead
    auto loadV = [&](Frag16& fr, int f) {
      const uint16_t* p = &Vtile[f * 16 + lr][hi ? 8 : 0];
      fr.q[0] = *(const uint4*)(p);
      fr.q[1] = *(const uint4*)(p + 16);
    };
    Frag16 vb;
    loadV(vb, 0);
#pragma unroll
    for (int f = 0; f < 8; ++f) {
      Frag16 vn;
      if (f + 1 < 8) loadV(vn, f + 1);
      acc[f] = wmma_bf16(pf, vb, acc[f]);
      if (f + 1 < 8) vb = vn;
    }
  }

  // normalize & store bf16
  float inv[8];
#pragma unroll
  for (int r = 0; r < 8; ++r) inv[r] = 1.0f / lrow[r];
#pragma unroll
  for (int f = 0; f < 8; ++f) {
    const int col = h * D + f * 16 + lr;
#pragma unroll
    for (int r = 0; r < 8; ++r) {
      const int qrow = q0 + r + (hi ? 8 : 0);
      Ob[(size_t)(b * S + qrow) * E + col] = f2bf(acc[f][r] * inv[r]);
    }
  }
}

// ---------------------------------------------------------------------------
// Host launch
// ---------------------------------------------------------------------------
extern "C" void kernel_launch(void* const* d_in, const int* in_sizes, int n_in,
                              void* d_out, int out_size, void* d_ws, size_t ws_size,
                              hipStream_t stream) {
  (void)in_sizes; (void)n_in; (void)out_size; (void)ws_size;
  constexpr int B = 4, S = 2048, E = 2048, H = 16, D = 128;
  constexpr size_t MR = (size_t)B * S;  // 8192 rows

  const float* x  = (const float*)d_in[0];
  const float* bq = (const float*)d_in[2];
  const float* bk = (const float*)d_in[4];
  const float* bv = (const float*)d_in[6];
  const float* bo = (const float*)d_in[8];

  // bf16 workspace layout (uint16 elements)
  uint16_t* ws = (uint16_t*)d_ws;
  size_t off = 0;
  uint16_t* xb  = ws + off; off += MR * E;        // 16,777,216
  uint16_t* Qb  = ws + off; off += MR * E;        // 16,777,216
  uint16_t* Kb  = ws + off; off += MR * D;        //  1,048,576
  uint16_t* Vtr = ws + off; off += MR * D;        //  1,048,576
  uint16_t* Ab  = ws + off; off += MR * E;        // 16,777,216
  uint16_t* Wqb = ws + off; off += (size_t)E * E; //  4,194,304
  uint16_t* Wkb = ws + off; off += (size_t)D * E; //    262,144
  uint16_t* Wvb = ws + off; off += (size_t)D * E; //    262,144
  uint16_t* Wob = ws + off; off += (size_t)E * E; //  4,194,304

  cvt_f32_to_bf16<<<4096, 256, 0, stream>>>(x, xb, MR * E);
  cvt_f32_to_bf16<<<2048, 256, 0, stream>>>((const float*)d_in[1], Wqb, (size_t)E * E);
  cvt_f32_to_bf16<<<512,  256, 0, stream>>>((const float*)d_in[3], Wkb, (size_t)D * E);
  cvt_f32_to_bf16<<<512,  256, 0, stream>>>((const float*)d_in[5], Wvb, (size_t)D * E);
  cvt_f32_to_bf16<<<2048, 256, 0, stream>>>((const float*)d_in[7], Wob, (size_t)E * E);

  // Q = xb @ Wq^T + bq          -> [8192, 2048] bf16
  gemm_bf16_wmma<2, 4, false, false><<<1024, 256, 0, stream>>>(
      xb, Wqb, bq, Qb, (int)MR, E, E, S);
  // K = xb @ Wk^T + bk          -> [8192, 128] bf16
  gemm_bf16_wmma<2, 4, false, false><<<64, 256, 0, stream>>>(
      xb, Wkb, bk, Kb, (int)MR, D, E, S);
  // V^T = (xb @ Wv^T + bv)^T    -> [4, 128, 2048] bf16
  gemm_bf16_wmma<2, 4, true, false><<<64, 256, 0, stream>>>(
      xb, Wvb, bv, Vtr, (int)MR, D, E, S);

  // fused MQA attention -> Ab [8192, 2048] bf16
  mqa_attn<<<B * H * (S / 128), 256, 0, stream>>>(Qb, Kb, Vtr, Ab);

  // out = Ab @ Wo^T + bo        -> [8192, 2048] fp32
  gemm_bf16_wmma<2, 4, false, true><<<1024, 256, 0, stream>>>(
      Ab, Wob, bo, d_out, (int)MR, E, E, S);
}
